// MambaTemporalProcessor_66133906424334
// MI455X (gfx1250) — compile-verified, matrix operand order checked by reading, not run on hardware
//
#include <hip/hip_runtime.h>
#include <hip/hip_bf16.h>

typedef __attribute__((ext_vector_type(16))) _Float16 v16h;
typedef __attribute__((ext_vector_type(8)))  float    v8f;

#define B_     256
#define L_     128
#define DM_    384
#define DI_    768
#define NH_    12
#define HD_    64
#define DSTATE_ 64
#define CONVD_ 896
#define DCONV_ 4
#define DIP_   1676      // 2*DI + 2*DSTATE + NH
#define DIPP_  1728      // padded to /64 for 4-wide N register blocking
#define NT_    (B_*L_)   // 32768 tokens
#define EPS_   1e-5f

// ---------------------------------------------------------------------------
// WMMA fragment loader: 16-bit A-matrix 16x32 layout (ISA 7.12.2):
//   lanes 0-15  (row = lane):    VGPR0-3 -> K=0..7,  VGPR4-7 -> K=16..23
//   lanes 16-31 (row = lane-16): VGPR0-3 -> K=8..15, VGPR4-7 -> K=24..31
// element e: k = e + (e>=8 ? 8 : 0) + laneHi*8.  B uses the symmetric mapping
// with N = laneLow.  Pairs are K-contiguous -> compiler merges to b128 loads.
// ---------------------------------------------------------------------------
union FragU { v16h h; unsigned u[8]; };

__device__ __forceinline__ v16h load_frag(const _Float16* __restrict__ base,
                                          int ld, int row, int k0, int laneHi) {
  const _Float16* p = base + (size_t)row * ld + k0 + laneHi * 8;
  FragU f;
#pragma unroll
  for (int q = 0; q < 8; ++q) {
    int off = (q < 4) ? 2 * q : 2 * q + 8;   // halves: 0,2,4,6, 16,18,20,22
    f.u[q] = *(const unsigned*)(p + off);
  }
  return f.h;
}

#define WMMA_F16(a, b, c) __builtin_amdgcn_wmma_f32_16x16x32_f16( \
    false, (a), false, (b), (short)0, (c), false, false)

// D(MxN) = A(MxK) * W(NxK)^T [+ resid]; f16 in, f32 accumulate.
// Register blocking: each wave owns a 32(M) x 64(N) super-tile:
// 2 A-frags x 4 B-frags -> 8 WMMAs per K-step, 1.5 b128 loads per WMMA.
__global__ __launch_bounds__(256) void gemm_wmma_f16(
    const _Float16* __restrict__ A, const _Float16* __restrict__ W,
    const float* __restrict__ resid, float* __restrict__ D,
    int K, int ldD, int nSuper, int totalSuper)
{
  int wave = threadIdx.x >> 5;
  int lane = threadIdx.x & 31;
  int st = blockIdx.x * 8 + wave;
  if (st >= totalSuper) return;               // wave-uniform: EXEC stays all-1
  int ns = st % nSuper;
  int ms = st / nSuper;
  int m0 = ms * 32, n0 = ns * 64;
  int laneLow = lane & 15, laneHi = lane >> 4;

  v8f acc[2][4];
#pragma unroll
  for (int i = 0; i < 2; ++i)
#pragma unroll
    for (int j = 0; j < 4; ++j) acc[i][j] = (v8f){};

  for (int k0 = 0; k0 < K; k0 += 32) {
    v16h a0 = load_frag(A, K, m0 +      laneLow, k0, laneHi);
    v16h a1 = load_frag(A, K, m0 + 16 + laneLow, k0, laneHi);
    v16h b0 = load_frag(W, K, n0 +      laneLow, k0, laneHi);
    v16h b1 = load_frag(W, K, n0 + 16 + laneLow, k0, laneHi);
    v16h b2 = load_frag(W, K, n0 + 32 + laneLow, k0, laneHi);
    v16h b3 = load_frag(W, K, n0 + 48 + laneLow, k0, laneHi);
    acc[0][0] = WMMA_F16(a0, b0, acc[0][0]);
    acc[0][1] = WMMA_F16(a0, b1, acc[0][1]);
    acc[0][2] = WMMA_F16(a0, b2, acc[0][2]);
    acc[0][3] = WMMA_F16(a0, b3, acc[0][3]);
    acc[1][0] = WMMA_F16(a1, b0, acc[1][0]);
    acc[1][1] = WMMA_F16(a1, b1, acc[1][1]);
    acc[1][2] = WMMA_F16(a1, b2, acc[1][2]);
    acc[1][3] = WMMA_F16(a1, b3, acc[1][3]);
  }

#pragma unroll
  for (int i = 0; i < 2; ++i) {
#pragma unroll
    for (int j = 0; j < 4; ++j) {
      int n = n0 + j * 16 + laneLow;
#pragma unroll
      for (int r = 0; r < 8; ++r) {            // C/D: VGPR r -> M=r (+8 hi half)
        int m = m0 + i * 16 + r + laneHi * 8;
        size_t idx = (size_t)m * ldD + n;
        float v = acc[i][j][r];
        if (resid) v += resid[idx];
        D[idx] = v;
      }
    }
  }
}

// ---------------------------------------------------------------------------
// Weight conversion (zero-pads in_proj rows 1676..1727)
// ---------------------------------------------------------------------------
__global__ void cvt_w1_kernel(const float* __restrict__ w, _Float16* __restrict__ wh, int n) {
  int i = blockIdx.x * blockDim.x + threadIdx.x;
  if (i >= n) return;
  int c = i % DM_;
  int r = (i / DM_) % DIPP_;
  int l = i / (DM_ * DIPP_);
  wh[i] = (r < DIP_) ? (_Float16)w[((size_t)l * DIP_ + r) * DM_ + c] : (_Float16)0.f;
}

__global__ void cvt_w2_kernel(const float* __restrict__ w, _Float16* __restrict__ wh, int n) {
  int i = blockIdx.x * blockDim.x + threadIdx.x;
  if (i >= n) return;
  wh[i] = (_Float16)w[i];
}

// ---------------------------------------------------------------------------
// x = emb + sinusoidal positional encoding
// ---------------------------------------------------------------------------
__global__ void pe_add_kernel(const float* __restrict__ emb, float* __restrict__ x, int n) {
  int i = blockIdx.x * blockDim.x + threadIdx.x;
  if (i >= n) return;
  int d = i % DM_;
  int t = (i / DM_) % L_;
  int i2 = d & ~1;
  float div = __expf((float)i2 * (-9.210340371976184f / (float)DM_)); // -ln(1e4)/d
  float ang = (float)t * div;
  float pe = (d & 1) ? __cosf(ang) : __sinf(ang);
  x[i] = emb[i] + pe;
}

// ---------------------------------------------------------------------------
// Per-token LayerNorm over 384 -> f16 (128 threads, 3 elems each)
// ---------------------------------------------------------------------------
__global__ __launch_bounds__(128) void ln_f16_kernel(
    const float* __restrict__ x, const float* __restrict__ w,
    const float* __restrict__ b, _Float16* __restrict__ out)
{
  __shared__ float s1[128], s2[128];
  int tok = blockIdx.x;
  const float* row = x + (size_t)tok * DM_;
  float v[3]; float sum = 0.f, sq = 0.f;
#pragma unroll
  for (int j = 0; j < 3; ++j) {
    int c = threadIdx.x + 128 * j;
    v[j] = row[c]; sum += v[j]; sq += v[j] * v[j];
  }
  s1[threadIdx.x] = sum; s2[threadIdx.x] = sq; __syncthreads();
  for (int o = 64; o > 0; o >>= 1) {
    if (threadIdx.x < o) { s1[threadIdx.x] += s1[threadIdx.x + o]; s2[threadIdx.x] += s2[threadIdx.x + o]; }
    __syncthreads();
  }
  float mu = s1[0] * (1.f / DM_);
  float rstd = rsqrtf(s2[0] * (1.f / DM_) - mu * mu + EPS_);
  _Float16* orow = out + (size_t)tok * DM_;
#pragma unroll
  for (int j = 0; j < 3; ++j) {
    int c = threadIdx.x + 128 * j;
    orow[c] = (_Float16)((v[j] - mu) * rstd * w[c] + b[c]);
  }
}

// ---------------------------------------------------------------------------
// dt = softplus(zxbcdt[..., 1664+h] + dt_bias[h])
// ---------------------------------------------------------------------------
__global__ void dt_kernel(const float* __restrict__ zxb, const float* __restrict__ dt_bias,
                          float* __restrict__ dt, int n) {
  int i = blockIdx.x * blockDim.x + threadIdx.x;
  if (i >= n) return;
  int h = i % NH_; int tok = i / NH_;
  float v = zxb[(size_t)tok * DIPP_ + (DI_ + CONVD_) + h] + dt_bias[h];
  dt[i] = (v > 20.f) ? v : log1pf(__expf(v));
}

// ---------------------------------------------------------------------------
// Causal depthwise conv (4 taps) + bias + SiLU over 896 channels
// ---------------------------------------------------------------------------
__global__ void conv_silu_kernel(const float* __restrict__ zxb, const float* __restrict__ cw,
                                 const float* __restrict__ cb, float* __restrict__ xbc, int n) {
  int i = blockIdx.x * blockDim.x + threadIdx.x;
  if (i >= n) return;
  int c = i % CONVD_; int tok = i / CONVD_; int t = tok % L_;
  float acc = cb[c];
#pragma unroll
  for (int k = 0; k < DCONV_; ++k) {
    int ts = t - (DCONV_ - 1) + k;
    if (ts >= 0)
      acc += zxb[(size_t)(tok - (DCONV_ - 1) + k) * DIPP_ + DI_ + c] * cw[c * DCONV_ + k];
  }
  xbc[i] = acc * (1.f / (1.f + __expf(-acc)));
}

// ---------------------------------------------------------------------------
// SSM scan: one block per (batch, head); 64x64 state in registers
// (16 f32/thread), LDS broadcast of x/B/C per step, 4-lane shfl_xor reduce.
// ---------------------------------------------------------------------------
__global__ __launch_bounds__(256) void scan_kernel(
    const float* __restrict__ xbc, const float* __restrict__ dtb,
    const float* __restrict__ A_log, const float* __restrict__ Dp,
    float* __restrict__ y)
{
  __shared__ float sx[HD_], sB[DSTATE_], sC[DSTATE_];
  int b = blockIdx.x / NH_;
  int h = blockIdx.x % NH_;
  float A = -__expf(A_log[h]);
  float Dh = Dp[h];
  int p  = threadIdx.x >> 2;          // 0..63 : HEADDIM row
  int nb = (threadIdx.x & 3) * 16;    // 16-wide slice of the 64 states

  float hreg[16];
#pragma unroll
  for (int j = 0; j < 16; ++j) hreg[j] = 0.f;

  for (int t = 0; t < L_; ++t) {
    size_t tok = (size_t)b * L_ + t;
    const float* row = xbc + tok * CONVD_;
    if (threadIdx.x < 64)        sx[threadIdx.x]       = row[h * HD_ + threadIdx.x];
    else if (threadIdx.x < 128)  sB[threadIdx.x - 64]  = row[DI_ + (threadIdx.x - 64)];
    else if (threadIdx.x < 192)  sC[threadIdx.x - 128] = row[DI_ + DSTATE_ + (threadIdx.x - 128)];
    __syncthreads();

    float dt = dtb[tok * NH_ + h];
    float dA = __expf(dt * A);
    float xp = sx[p];
    float coef = dt * xp;
    float yp = 0.f;
#pragma unroll
    for (int j = 0; j < 16; ++j) {
      float Bn = sB[nb + j], Cn = sC[nb + j];
      hreg[j] = hreg[j] * dA + coef * Bn;
      yp += hreg[j] * Cn;
    }
    yp += __shfl_xor(yp, 1);
    yp += __shfl_xor(yp, 2);
    if ((threadIdx.x & 3) == 0)
      y[tok * DI_ + h * HD_ + p] = yp + Dh * xp;
    __syncthreads();
  }
}

// ---------------------------------------------------------------------------
// g = y * silu(z); RMSNorm over 768 with weight; output f16 for GEMM2
// ---------------------------------------------------------------------------
__global__ __launch_bounds__(256) void gate_rms_kernel(
    const float* __restrict__ y, const float* __restrict__ zxb,
    const float* __restrict__ nw, _Float16* __restrict__ gh)
{
  __shared__ float sred[256];
  int tok = blockIdx.x;
  const float* yrow = y + (size_t)tok * DI_;
  const float* zrow = zxb + (size_t)tok * DIPP_;   // z = cols [0,768)
  float g[3]; float sq = 0.f;
#pragma unroll
  for (int j = 0; j < 3; ++j) {
    int c = threadIdx.x + 256 * j;
    float z = zrow[c];
    float gv = yrow[c] * (z / (1.f + __expf(-z)));
    g[j] = gv; sq += gv * gv;
  }
  sred[threadIdx.x] = sq; __syncthreads();
  for (int o = 128; o > 0; o >>= 1) {
    if (threadIdx.x < o) sred[threadIdx.x] += sred[threadIdx.x + o];
    __syncthreads();
  }
  float r = rsqrtf(sred[0] * (1.f / DI_) + EPS_);
  _Float16* grow = gh + (size_t)tok * DI_;
#pragma unroll
  for (int j = 0; j < 3; ++j) {
    int c = threadIdx.x + 256 * j;
    grow[c] = (_Float16)(g[j] * r * nw[c]);
  }
}

// ---------------------------------------------------------------------------
// Final LayerNorm on last token + 384->60 projection
// ---------------------------------------------------------------------------
__global__ __launch_bounds__(128) void head_kernel(
    const float* __restrict__ x, const float* __restrict__ w, const float* __restrict__ b,
    const float* __restrict__ pw, const float* __restrict__ pb, float* __restrict__ out)
{
  __shared__ float s1[128], s2[128], sn[DM_];
  int bi = blockIdx.x;
  const float* row = x + ((size_t)bi * L_ + (L_ - 1)) * DM_;
  float v[3]; float sum = 0.f, sq = 0.f;
#pragma unroll
  for (int j = 0; j < 3; ++j) {
    int c = threadIdx.x + 128 * j;
    v[j] = row[c]; sum += v[j]; sq += v[j] * v[j];
  }
  s1[threadIdx.x] = sum; s2[threadIdx.x] = sq; __syncthreads();
  for (int o = 64; o > 0; o >>= 1) {
    if (threadIdx.x < o) { s1[threadIdx.x] += s1[threadIdx.x + o]; s2[threadIdx.x] += s2[threadIdx.x + o]; }
    __syncthreads();
  }
  float mu = s1[0] * (1.f / DM_);
  float rstd = rsqrtf(s2[0] * (1.f / DM_) - mu * mu + EPS_);
#pragma unroll
  for (int j = 0; j < 3; ++j) {
    int c = threadIdx.x + 128 * j;
    sn[c] = (v[j] - mu) * rstd * w[c] + b[c];
  }
  __syncthreads();
  if (threadIdx.x < 60) {
    float acc = pb[threadIdx.x];
    const float* pr = pw + threadIdx.x * DM_;
    for (int d = 0; d < DM_; ++d) acc += sn[d] * pr[d];
    out[bi * 60 + threadIdx.x] = acc;
  }
}

// ---------------------------------------------------------------------------
extern "C" void kernel_launch(void* const* d_in, const int* in_sizes, int n_in,
                              void* d_out, int out_size, void* d_ws, size_t ws_size,
                              hipStream_t stream) {
  (void)in_sizes; (void)n_in; (void)out_size; (void)ws_size;
  const float* emb        = (const float*)d_in[0];
  const float* in_proj_w  = (const float*)d_in[1];
  const float* conv_w     = (const float*)d_in[2];
  const float* conv_b     = (const float*)d_in[3];
  const float* dt_bias    = (const float*)d_in[4];
  const float* A_log      = (const float*)d_in[5];
  const float* Dparam     = (const float*)d_in[6];
  const float* mamba_nw   = (const float*)d_in[7];
  const float* out_proj_w = (const float*)d_in[8];
  const float* ln_w       = (const float*)d_in[9];
  const float* ln_b       = (const float*)d_in[10];
  const float* out_norm_w = (const float*)d_in[11];
  const float* out_norm_b = (const float*)d_in[12];
  const float* proj_w     = (const float*)d_in[13];
  const float* proj_b     = (const float*)d_in[14];
  float* out = (float*)d_out;

  // carve scratch (256B aligned)
  char* ws = (char*)d_ws;
  size_t off = 0;
  auto carve = [&](size_t bytes) -> char* {
    char* p = ws + off;
    off = (off + bytes + 255) & ~(size_t)255;
    return p;
  };
  float*     x    = (float*)    carve((size_t)NT_ * DM_ * 4);     // residual stream
  _Float16*  xnh  = (_Float16*) carve((size_t)NT_ * DM_ * 2);     // LN output f16
  float*     zxb  = (float*)    carve((size_t)NT_ * DIPP_ * 4);   // in_proj out (padded)
  float*     xbc  = (float*)    carve((size_t)NT_ * CONVD_ * 4);  // conv+silu out
  float*     dtb  = (float*)    carve((size_t)NT_ * NH_ * 4);     // softplus dt
  float*     ybuf = (float*)    carve((size_t)NT_ * DI_ * 4);     // scan out
  _Float16*  gh   = (_Float16*) carve((size_t)NT_ * DI_ * 2);     // gated/normed f16
  _Float16*  w1h  = (_Float16*) carve((size_t)2 * DIPP_ * DM_ * 2);
  _Float16*  w2h  = (_Float16*) carve((size_t)2 * DM_ * DI_ * 2);

  { int n = 2 * DIPP_ * DM_; cvt_w1_kernel<<<(n + 255) / 256, 256, 0, stream>>>(in_proj_w, w1h, n); }
  { int n = 2 * DM_ * DI_;   cvt_w2_kernel<<<(n + 255) / 256, 256, 0, stream>>>(out_proj_w, w2h, n); }
  { int n = NT_ * DM_;       pe_add_kernel<<<(n + 255) / 256, 256, 0, stream>>>(emb, x, n); }

  const int mSuper = NT_ / 32;                 // 1024 super-rows of 32

  for (int l = 0; l < 2; ++l) {
    ln_f16_kernel<<<NT_, 128, 0, stream>>>(x, ln_w + l * DM_, ln_b + l * DM_, xnh);

    { // in_proj: M=NT, N=1728(pad), K=384 ; 32x64 super-tiles
      int nSuper = DIPP_ / 64;                 // 27
      int total  = mSuper * nSuper;            // 27648
      gemm_wmma_f16<<<(total + 7) / 8, 256, 0, stream>>>(
          xnh, w1h + (size_t)l * DIPP_ * DM_, nullptr, zxb, DM_, DIPP_, nSuper, total);
    }
    { int n = NT_ * NH_;    dt_kernel<<<(n + 255) / 256, 256, 0, stream>>>(zxb, dt_bias + l * NH_, dtb, n); }
    { int n = NT_ * CONVD_; conv_silu_kernel<<<(n + 255) / 256, 256, 0, stream>>>(
          zxb, conv_w + l * CONVD_ * DCONV_, conv_b + l * CONVD_, xbc, n); }

    scan_kernel<<<B_ * NH_, 256, 0, stream>>>(xbc, dtb, A_log + l * NH_, Dparam + l * NH_, ybuf);

    gate_rms_kernel<<<NT_, 256, 0, stream>>>(ybuf, zxb, mamba_nw + l * DI_, gh);

    { // out_proj + residual (in-place on x; each element read-once/write-once)
      int nSuper = DM_ / 64;                   // 6
      int total  = mSuper * nSuper;            // 6144
      gemm_wmma_f16<<<(total + 7) / 8, 256, 0, stream>>>(
          gh, w2h + (size_t)l * DM_ * DI_, x, x, DI_, DM_, nSuper, total);
    }
  }

  head_kernel<<<B_, 128, 0, stream>>>(x, out_norm_w, out_norm_b, proj_w, proj_b, out);
}
